// CrossDomainBridge_17394617548905
// MI455X (gfx1250) — matrix operand order repaired and checked by reading, not compiled, not run on hardware
//
#include <hip/hip_runtime.h>
#include <hip/hip_bf16.h>

// ---- problem dims (fixed by reference) ----
#define BB 8
#define NN 2048
#define MM (BB * NN)   // 16384 rows
#define CC 64
#define AA 256
#define FF 128

// ---- output layout (floats) ----
#define OUT_CORR 2097152            // 8*2048*128
#define OUT_DW   2097224            // + 72
#define OUT_ENH  2097248            // + 24

// ---- workspace layout (bytes) ----
#define WS_WT_S   0u                //  WsT  [256][64]  f16
#define WS_WT_F   32768u
#define WS_WT_W   65536u
#define WS_WT_T   98304u            //  WtT  [128][256] f16
#define WS_WT_G   163840u           //  WgT  [128][128] f16
#define WS_WT_FU  196608u           //  WfuT [128][384] f16
#define WS_WT_DF  294912u           //  WdfT [128][128] f16
#define WS_CG     327680u           //  cg   [128] f32
#define WS_ALIGNED 331776u          //  3 x [16384][256] f16
#define WS_EBUF   25497600u         //  3 x [16384][128] f16

typedef __attribute__((ext_vector_type(16))) _Float16 v16h;
typedef __attribute__((ext_vector_type(8)))  _Float16 v8h;
typedef __attribute__((ext_vector_type(8)))  float    v8f;

static __device__ __forceinline__ v8f wmma_f16(v16h a, v16h b, v8f c) {
  // D = A(16x32 f16) * B(32x16 f16) + C(16x16 f32)
  return __builtin_amdgcn_wmma_f32_16x16x32_f16(false, a, false, b, (short)0, c,
                                                false, false);
}

// A-fragment loader: p = row base + kb*32 (f16 elements), hf = lane>=16
// a[0..7]  = A[m][kb*32 + hf*16 + 0..7]
// a[8..15] = A[m][kb*32 + hf*16 + 8 + 0..7]
static __device__ __forceinline__ v16h load_a(const _Float16* p, unsigned hf) {
  v8h lo = *(const v8h*)(p + hf * 16);
  v8h hi = *(const v8h*)(p + hf * 16 + 8);
  return __builtin_shufflevector(lo, hi, 0, 1, 2, 3, 4, 5, 6, 7, 8, 9, 10, 11,
                                 12, 13, 14, 15);
}

// ============================ kernel 0: prep ============================
__global__ __launch_bounds__(256) void prep_kernel(
    const float* Ws, const float* Wf, const float* Ww, const float* Wt,
    const float* Wg, const float* bg, const float* Wfu, const float* Wdf,
    _Float16* TWs, _Float16* TWf, _Float16* TWw, _Float16* TWt, _Float16* TWg,
    _Float16* TWfu, _Float16* TWdf, float* cg, float* out) {
  int tid = blockIdx.x * blockDim.x + threadIdx.x;
  int stride = gridDim.x * blockDim.x;
  for (int i = tid; i < 64 * 256; i += stride) {   // [64][256] -> [256][64]
    int k = i >> 8, n = i & 255;
    TWs[n * 64 + k] = (_Float16)Ws[i];
    TWf[n * 64 + k] = (_Float16)Wf[i];
    TWw[n * 64 + k] = (_Float16)Ww[i];
  }
  for (int i = tid; i < 256 * 128; i += stride) {  // [256][128] -> [128][256]
    int k = i >> 7, n = i & 127;
    TWt[n * 256 + k] = (_Float16)Wt[i];
  }
  for (int i = tid; i < 128 * 128; i += stride) {  // [128][128] transposes
    int k = i >> 7, n = i & 127;
    TWg[n * 128 + k]  = (_Float16)Wg[i];           // first 128 rows of Wg
    TWdf[n * 128 + k] = (_Float16)Wdf[i];
  }
  for (int i = tid; i < 384 * 128; i += stride) {  // [384][128] -> [128][384]
    int k = i >> 7, n = i & 127;
    TWfu[n * 384 + k] = (_Float16)Wfu[i];
  }
  if (tid < 128)  // fold constant corr_mean (=1/2048) rows of Wg into bias
    cg[tid] = bg[tid] + (Wg[128 * 128 + tid] + Wg[129 * 128 + tid] +
                         Wg[130 * 128 + tid]) * (1.0f / 2048.0f);
  if (tid < 72) out[OUT_CORR + tid] = 1.0f / 2048.0f;  // corr == 1/N
  if (tid >= 72 && tid < 96) out[OUT_DW + (tid - 72)] = 0.125f;  // dw == 1/B
}

// ======================= kernel 1: align + LN ==========================
// grid: (MM/64, 3), block 256.  aligned_d = LN(x_d @ W_d + b_d) -> f16
__global__ __launch_bounds__(256) void align_kernel(
    const float* x0, const float* x1, const float* x2, const _Float16* T0,
    const _Float16* T1, const _Float16* T2, const float* bb0, const float* bb1,
    const float* bb2, const float* gam, const float* bet, _Float16* alignedb) {
  __shared__ alignas(16) float sMem[64 * 256];     // 64 KB overlay region
  float* sXf = sMem;                               // [64][64] f32 (16 KB)
  _Float16* sXh = (_Float16*)(sMem + 4096);        // [64][64] f16 (8 KB)
  float* sY = sMem;                                // [64][256] f32 (later)

  int d = blockIdx.y;
  const float* X = (d == 0) ? x0 : (d == 1) ? x1 : x2;
  const _Float16* WT = (d == 0) ? T0 : (d == 1) ? T1 : T2;
  const float* bias = (d == 0) ? bb0 : (d == 1) ? bb1 : bb2;
  _Float16* dst = alignedb + (size_t)d * ((size_t)MM * AA);

  int rowbase = blockIdx.x * 64;
  unsigned t = threadIdx.x, lane = t & 31, w = t >> 5;
  unsigned hf = lane >> 4, l16 = lane & 15;

  // ---- async global->LDS copy of the contiguous 16 KB X tile ----
  unsigned ldsbase =
      (unsigned)(size_t)(__attribute__((address_space(3))) float*)sMem;
  const float* gp = X + (size_t)rowbase * 64;
#pragma unroll
  for (int it = 0; it < 4; ++it) {
    unsigned e = it * 256 + t;                     // 16-byte chunk id
    unsigned loff = ldsbase + e * 16;
    unsigned long long ga = (unsigned long long)(gp + e * 4);
    asm volatile("global_load_async_to_lds_b128 %0, %1, off"
                 :: "v"(loff), "v"(ga) : "memory");
  }
  asm volatile("s_wait_asynccnt 0x0" ::: "memory");
  // convert this wave's own chunks f32 -> f16 (wave-local dependency only)
#pragma unroll
  for (int it = 0; it < 4; ++it) {
    unsigned e = it * 256 + t;
#pragma unroll
    for (int j = 0; j < 4; ++j) sXh[e * 4 + j] = (_Float16)sXf[e * 4 + j];
  }
  __syncthreads();

  unsigned rt = w >> 1, ch = w & 1;                // wave: 16 rows x 128 cols
  const _Float16* arow = sXh + (rt * 16 + l16) * 64;
  v16h a0 = load_a(arow, hf);
  v16h a1 = load_a(arow + 32, hf);

  // preload all 16 B fragments (one big clause, no per-WMMA drain)
  v16h bfr[16];
#pragma unroll
  for (int ct = 0; ct < 8; ++ct) {
    const _Float16* bp = WT + (ch * 128 + ct * 16 + l16) * 64 + hf * 16;
    bfr[2 * ct]     = *(const v16h*)bp;
    bfr[2 * ct + 1] = *(const v16h*)(bp + 32);
  }
  __syncthreads();                                 // sX dead; sY may be written

#pragma unroll
  for (int ct = 0; ct < 8; ++ct) {
    int n = ch * 128 + ct * 16 + l16;
    v8f acc = {};
    acc = wmma_f16(a0, bfr[2 * ct], acc);
    acc = wmma_f16(a1, bfr[2 * ct + 1], acc);
    float bn = bias[n];
#pragma unroll
    for (int r = 0; r < 8; ++r)
      sY[(rt * 16 + r + 8 * hf) * 256 + n] = acc[r] + bn;
  }
  __syncthreads();

  // LayerNorm over 256 features; wave handles 8 rows, wave32 shuffle reduce
  for (int rr = 0; rr < 8; ++rr) {
    int row = w * 8 + rr;
    const float* yp = sY + row * 256 + lane * 8;
    float vals[8], s = 0.f, ss = 0.f;
#pragma unroll
    for (int j = 0; j < 8; ++j) {
      float v = yp[j]; vals[j] = v; s += v; ss += v * v;
    }
#pragma unroll
    for (int off = 16; off > 0; off >>= 1) {
      s += __shfl_xor(s, off, 32);
      ss += __shfl_xor(ss, off, 32);
    }
    float mean = s * (1.f / 256.f);
    float var = ss * (1.f / 256.f) - mean * mean;
    float inv = rsqrtf(var + 1e-3f);
    v8h o;
#pragma unroll
    for (int j = 0; j < 8; ++j) {
      int c = lane * 8 + j;
      o[j] = (_Float16)(gam[c] * (vals[j] - mean) * inv + bet[c]);
    }
    *(v8h*)(dst + (size_t)(rowbase + row) * 256 + lane * 8) = o;
  }
}

// ================== kernel 2: t, gate, enhanced ========================
// grid: (MM/128, 3), block 256. Each wave: 16 rows x full F=128.
__global__ __launch_bounds__(256) void transfer_kernel(
    const _Float16* alignedb, const _Float16* TWt, const _Float16* TWg,
    const float* bt, const float* cg, _Float16* eb, float* out) {
  __shared__ alignas(16) _Float16 sT[8][16 * 128];  // 32 KB, wave-private
  int d = blockIdx.y;
  const _Float16* A = alignedb + (size_t)d * ((size_t)MM * AA);
  _Float16* eh = eb + (size_t)d * ((size_t)MM * FF);
  float* eo = out + OUT_ENH + (size_t)d * ((size_t)MM * FF);

  unsigned t = threadIdx.x, lane = t & 31, w = t >> 5;
  unsigned hf = lane >> 4, l16 = lane & 15;
  int rbw = blockIdx.x * 128 + w * 16;

  // t = aligned @ Wt + bt   (K = 256)
  const _Float16* arow = A + (size_t)(rbw + l16) * 256;
  v16h a[8];
#pragma unroll
  for (int kb = 0; kb < 8; ++kb) a[kb] = load_a(arow + kb * 32, hf);

  auto loadBt = [&](v16h* dstf, int ct) {
    const _Float16* bp = TWt + (ct * 16 + l16) * 256 + hf * 16;
#pragma unroll
    for (int kb = 0; kb < 8; ++kb) dstf[kb] = *(const v16h*)(bp + kb * 32);
  };

  v16h bA[8], bB[8];
  loadBt(bA, 0);
  v8f tacc[8];
#pragma unroll
  for (int ct = 0; ct < 8; ++ct) {                 // double-buffered B
    v16h* bc = (ct & 1) ? bB : bA;
    v16h* bn = (ct & 1) ? bA : bB;
    if (ct < 7) loadBt(bn, ct + 1);
    int n = ct * 16 + l16;
    v8f acc = {};
#pragma unroll
    for (int kb = 0; kb < 8; ++kb) acc = wmma_f16(a[kb], bc[kb], acc);
    float bnv = bt[n];
#pragma unroll
    for (int r = 0; r < 8; ++r) {
      float tv = acc[r] + bnv;
      acc[r] = tv;
      sT[w][(r + 8 * hf) * 128 + n] = (_Float16)tv;
    }
    tacc[ct] = acc;
  }

  // gate = sigmoid(t @ Wg0 + cg)   (K = 128); e = t * gate * (1/8)
  const _Float16* trow = sT[w] + l16 * 128;
  v16h a2[4];
#pragma unroll
  for (int kb = 0; kb < 4; ++kb) a2[kb] = load_a(trow + kb * 32, hf);

  auto loadBg = [&](v16h* dstf, int ct) {
    const _Float16* bp = TWg + (ct * 16 + l16) * 128 + hf * 16;
#pragma unroll
    for (int kb = 0; kb < 4; ++kb) dstf[kb] = *(const v16h*)(bp + kb * 32);
  };

  v16h gA[4], gB[4];
  loadBg(gA, 0);
#pragma unroll
  for (int ct = 0; ct < 8; ++ct) {
    v16h* bc = (ct & 1) ? gB : gA;
    v16h* bn = (ct & 1) ? gA : gB;
    if (ct < 7) loadBg(bn, ct + 1);
    int n = ct * 16 + l16;
    v8f acc = {};
#pragma unroll
    for (int kb = 0; kb < 4; ++kb) acc = wmma_f16(a2[kb], bc[kb], acc);
    float cgn = cg[n];
#pragma unroll
    for (int r = 0; r < 8; ++r) {
      float gate = 1.0f / (1.0f + __expf(-(acc[r] + cgn)));
      float e = tacc[ct][r] * gate * 0.125f;
      size_t idx = (size_t)(rbw + r + 8 * hf) * 128 + n;
      eo[idx] = e;
      eh[idx] = (_Float16)e;
    }
  }
}

// ==================== kernel 3: fuse + final LN ========================
// grid: (MM/128), block 256. Each wave: 16 rows x full F=128.
__global__ __launch_bounds__(256) void fuse_kernel(
    const _Float16* eb, const _Float16* TWfu, const _Float16* TWdf,
    const float* bfu, const float* bdf, const float* g2, const float* be2,
    float* out) {
  __shared__ alignas(16) float sBuf[8][16 * 128];  // 64 KB, wave-private
  unsigned t = threadIdx.x, lane = t & 31, w = t >> 5;
  unsigned hf = lane >> 4, l16 = lane & 15;
  int rbw = blockIdx.x * 128 + w * 16;
  _Float16* sF = (_Float16*)sBuf[w];               // fused f16 (first 4 KB)
  float* sY = sBuf[w];                             // y f32 (after sF is dead)

  // fused = concat(e0,e1,e2) @ Wfu + bfu   (K = 384)
  v16h ae[12];
#pragma unroll
  for (int dkb = 0; dkb < 12; ++dkb) {
    int d = dkb >> 2, kb = dkb & 3;
    const _Float16* arow =
        eb + (size_t)d * ((size_t)MM * FF) + (size_t)(rbw + l16) * 128;
    ae[dkb] = load_a(arow + kb * 32, hf);
  }

  auto loadBfu = [&](v16h* dstf, int ct) {
    const _Float16* bp = TWfu + (ct * 16 + l16) * 384 + hf * 16;
#pragma unroll
    for (int dkb = 0; dkb < 12; ++dkb)
      dstf[dkb] = *(const v16h*)(bp + dkb * 32);   // d*128 + kb*32 == dkb*32
  };

  v16h fA[12], fB[12];
  loadBfu(fA, 0);
#pragma unroll
  for (int ct = 0; ct < 8; ++ct) {
    v16h* bc = (ct & 1) ? fB : fA;
    v16h* bn = (ct & 1) ? fA : fB;
    if (ct < 7) loadBfu(bn, ct + 1);
    int n = ct * 16 + l16;
    v8f acc = {};
#pragma unroll
    for (int dkb = 0; dkb < 12; ++dkb) acc = wmma_f16(ae[dkb], bc[dkb], acc);
    float bnv = bfu[n];
#pragma unroll
    for (int r = 0; r < 8; ++r)
      sF[(r + 8 * hf) * 128 + n] = (_Float16)(acc[r] + bnv);
  }

  // y = fused @ Wdf + bdf   (K = 128)
  const _Float16* frow = sF + l16 * 128;
  v16h a3[4];
#pragma unroll
  for (int kb = 0; kb < 4; ++kb) a3[kb] = load_a(frow + kb * 32, hf);

  auto loadBdf = [&](v16h* dstf, int ct) {
    const _Float16* bp = TWdf + (ct * 16 + l16) * 128 + hf * 16;
#pragma unroll
    for (int kb = 0; kb < 4; ++kb) dstf[kb] = *(const v16h*)(bp + kb * 32);
  };

  v16h dA[4], dB[4];
  loadBdf(dA, 0);
#pragma unroll
  for (int ct = 0; ct < 8; ++ct) {
    v16h* bc = (ct & 1) ? dB : dA;
    v16h* bn = (ct & 1) ? dA : dB;
    if (ct < 7) loadBdf(bn, ct + 1);
    int n = ct * 16 + l16;
    v8f acc = {};
#pragma unroll
    for (int kb = 0; kb < 4; ++kb) acc = wmma_f16(a3[kb], bc[kb], acc);
    float bnv = bdf[n];
#pragma unroll
    for (int r = 0; r < 8; ++r)
      sY[(r + 8 * hf) * 128 + n] = acc[r] + bnv;
  }

  // final LayerNorm over 128 features
  for (int row = 0; row < 16; ++row) {
    const float* yp = sY + row * 128 + lane * 4;
    float vals[4], s = 0.f, ss = 0.f;
#pragma unroll
    for (int j = 0; j < 4; ++j) {
      float v = yp[j]; vals[j] = v; s += v; ss += v * v;
    }
#pragma unroll
    for (int off = 16; off > 0; off >>= 1) {
      s += __shfl_xor(s, off, 32);
      ss += __shfl_xor(ss, off, 32);
    }
    float mean = s * (1.f / 128.f);
    float var = ss * (1.f / 128.f) - mean * mean;
    float inv = rsqrtf(var + 1e-3f);
    float4 o;
#pragma unroll
    for (int j = 0; j < 4; ++j) {
      int c = lane * 4 + j;
      ((float*)&o)[j] = g2[c] * (vals[j] - mean) * inv + be2[c];
    }
    *(float4*)(out + (size_t)(rbw + row) * 128 + lane * 4) = o;
  }
}

extern "C" void kernel_launch(void* const* d_in, const int* in_sizes, int n_in,
                              void* d_out, int out_size, void* d_ws,
                              size_t ws_size, hipStream_t stream) {
  (void)in_sizes; (void)n_in; (void)out_size; (void)ws_size;
  const float* spatial   = (const float*)d_in[0];
  const float* frequency = (const float*)d_in[1];
  const float* wavelet   = (const float*)d_in[2];
  const float* Ws  = (const float*)d_in[3];
  const float* bs  = (const float*)d_in[4];
  const float* Wf  = (const float*)d_in[5];
  const float* bf  = (const float*)d_in[6];
  const float* Ww  = (const float*)d_in[7];
  const float* bw  = (const float*)d_in[8];
  const float* g1  = (const float*)d_in[9];
  const float* b1  = (const float*)d_in[10];
  // d_in[11..14] = Wk, bk, Wq, bq: unused (attention means are analytic 1/N)
  const float* Wt  = (const float*)d_in[15];
  const float* bt  = (const float*)d_in[16];
  const float* Wg  = (const float*)d_in[17];
  const float* bg  = (const float*)d_in[18];
  const float* Wfu = (const float*)d_in[19];
  const float* bfu = (const float*)d_in[20];
  const float* Wdf = (const float*)d_in[21];
  const float* bdf = (const float*)d_in[22];
  const float* g2  = (const float*)d_in[23];
  const float* b2  = (const float*)d_in[24];
  float* out = (float*)d_out;

  char* ws = (char*)d_ws;
  _Float16* TWs  = (_Float16*)(ws + WS_WT_S);
  _Float16* TWf  = (_Float16*)(ws + WS_WT_F);
  _Float16* TWw  = (_Float16*)(ws + WS_WT_W);
  _Float16* TWt  = (_Float16*)(ws + WS_WT_T);
  _Float16* TWg  = (_Float16*)(ws + WS_WT_G);
  _Float16* TWfu = (_Float16*)(ws + WS_WT_FU);
  _Float16* TWdf = (_Float16*)(ws + WS_WT_DF);
  float*    cg   = (float*)(ws + WS_CG);
  _Float16* alignedb = (_Float16*)(ws + WS_ALIGNED);
  _Float16* eb       = (_Float16*)(ws + WS_EBUF);

  prep_kernel<<<64, 256, 0, stream>>>(Ws, Wf, Ww, Wt, Wg, bg, Wfu, Wdf, TWs,
                                      TWf, TWw, TWt, TWg, TWfu, TWdf, cg, out);
  align_kernel<<<dim3(MM / 64, 3), 256, 0, stream>>>(
      spatial, frequency, wavelet, TWs, TWf, TWw, bs, bf, bw, g1, b1, alignedb);
  transfer_kernel<<<dim3(MM / 128, 3), 256, 0, stream>>>(alignedb, TWt, TWg,
                                                         bt, cg, eb, out);
  fuse_kernel<<<dim3(MM / 128), 256, 0, stream>>>(eb, TWfu, TWdf, bfu, bdf, g2,
                                                  b2, out);
}
